// GlobalMultiHeadAttentionLayer_22144851378316
// MI455X (gfx1250) — matrix operand order repaired and checked by reading, not compiled, not run on hardware
//
#include <hip/hip_runtime.h>
#include <math.h>
#include <stdint.h>

// ---------------- types for CDNA5 WMMA ----------------
typedef __bf16 bf16;
typedef __attribute__((ext_vector_type(16))) __bf16 v16bf;
typedef __attribute__((ext_vector_type(8)))  __bf16 v8bf;
typedef __attribute__((ext_vector_type(8)))  float  v8f;

// ---------------- problem constants ----------------
#define B_   2
#define L_   1024
#define D_   1024
#define H_   16
#define DK_  64
#define PK_  5
#define LD_  ((long)B_ * L_ * D_)        // 2,097,152 elems
#define ZN_  (B_ * H_)                   // 32 (b,h) batches
#define SQ_  ((long)L_ * L_)             // 1,048,576 per-(b,h) attn tile
#define VT_  ((long)DK_ * L_)            // 65,536 per-(b,h) V^T tile
#define ATT_ ((long)ZN_ * SQ_)           // 33,554,432 elems per attn map

// d_out regions (floats): x | as_s | am_s | as_c | am_c
#define OFF_X    0L
#define OFF_ASS  (OFF_X + LD_)
#define OFF_AMS  (OFF_ASS + ATT_)
#define OFF_ASC  (OFF_AMS + ATT_)
#define OFF_AMC  (OFF_ASC + ATT_)

// LDS tile geometry: 128 rows x 32 K bf16, rows padded to 40 elems (80B) for banking
#define TROW_    40
#define TILE_B_  (128 * TROW_ * 2)       // 10240 bytes per buffer

// ---- CDNA5 async global->LDS copy (ASYNCcnt-tracked, no VGPR round trip) ----
__device__ __forceinline__ void async_copy_b128(unsigned lds_off, const void* gptr) {
    asm volatile("global_load_async_to_lds_b128 %0, %1, off"
                 :: "v"(lds_off), "v"(gptr) : "memory");
}
__device__ __forceinline__ void wait_async0() {
    asm volatile("s_wait_asynccnt 0x0" ::: "memory");
}

// =====================================================================
// Generic batched WMMA GEMM:  C[z][M,N] = act(A[z][M,K] @ W[z][N,K]^T + bias) (+resid)
// A,W row-major with K contiguous. Batch offset = (z/zInner)*Outer + (z%zInner)*Inner.
// Block = 256 threads = 8 waves laid out 4(M) x 2(N); wave tile 32x64
// (8 v_wmma_f32_16x16x32_bf16 per K=32 step). K slabs are double-buffered
// through LDS with global_load_async_to_lds_b128; fragments come from ds_load.
// =====================================================================
__global__ __launch_bounds__(256) void gemm_bf16_kernel(
    const bf16* __restrict__ A, const bf16* __restrict__ W,
    float* __restrict__ C, bf16* __restrict__ Cb,
    const float* __restrict__ bias, const float* __restrict__ resid,
    int M, int N, int K, int lda, int ldb, int ldc,
    long aOuter, long aInner, long wOuter, long wInner,
    long cOuter, long cInner, int zInner, int act)
{
    __shared__ bf16 sA[2][128][TROW_];
    __shared__ bf16 sB[2][128][TROW_];

    const int z    = blockIdx.z;
    const int wid  = threadIdx.x >> 5;
    const int lane = threadIdx.x & 31;
    const int half = lane >> 4;          // 0: lanes 0-15, 1: lanes 16-31
    const int l16  = lane & 15;
    const int wm   = wid & 3;
    const int wn   = wid >> 2;
    const int mBase = blockIdx.y * 128;
    const int nBase = blockIdx.x * 128;
    const int m0 = mBase + wm * 32;
    const int n0 = nBase + wn * 64;
    const bool active = (n0 < N);        // wave-uniform; inactive waves still copy+barrier

    const long aOff = (long)(z / zInner) * aOuter + (long)(z % zInner) * aInner;
    const long wOff = (long)(z / zInner) * wOuter + (long)(z % zInner) * wInner;
    const long cOff = (long)(z / zInner) * cOuter + (long)(z % zInner) * cInner;
    A += aOff; W += wOff;

    const unsigned sAoff = (unsigned)(uintptr_t)(&sA[0][0][0]);
    const unsigned sBoff = (unsigned)(uintptr_t)(&sB[0][0][0]);

    // stage one 128x32 A slab + 128x32 B slab into LDS buffer `buf`
    auto issueTile = [&](int buf, int kc) {
        #pragma unroll
        for (int it = 0; it < 4; it++) {
            const int c = threadIdx.x + it * 256;        // 0..1023 chunk id
            const int cc  = c & 511;
            const int row = cc >> 2;                     // 0..127
            const int kb  = (cc & 3) * 8;                // element offset in slab
            if (c < 512) {
                const bf16* g = A + (long)(mBase + row) * lda + (kc + kb);
                async_copy_b128(sAoff + buf * TILE_B_ + (row * TROW_ + kb) * 2, g);
            } else {
                int wrow = nBase + row; if (wrow > N - 1) wrow = N - 1;   // clamp for N<128
                const bf16* g = W + (long)wrow * ldb + (kc + kb);
                async_copy_b128(sBoff + buf * TILE_B_ + (row * TROW_ + kb) * 2, g);
            }
        }
    };

    v8f acc[2][4];
    #pragma unroll
    for (int i = 0; i < 2; i++)
        #pragma unroll
        for (int j = 0; j < 4; j++)
            #pragma unroll
            for (int r = 0; r < 8; r++) acc[i][j][r] = 0.f;

    issueTile(0, 0);
    int ib = 0;
    for (int kc = 0; kc < K; kc += 32, ib ^= 1) {
        wait_async0();          // this wave's slab copies landed in LDS
        __syncthreads();        // everyone's landed; prior reads of buf ib done
        if (kc + 32 < K) issueTile(ib ^ 1, kc + 32);   // stream next slab during math

        if (active) {
            // ---- A fragments (ISA 16x32 layout):
            // lane<16: elems 0..7 = K+0..7, 8..15 = K+16..23; lane>=16: +8 / +24
            const bf16* aT = &sA[ib][0][0];
            const bf16* bT = &sB[ib][0][0];
            v16bf af[2];
            #pragma unroll
            for (int i = 0; i < 2; i++) {
                const bf16* ap = aT + (wm * 32 + i * 16 + l16) * TROW_ + half * 8;
                union { v16bf v; v8bf h[2]; } u;
                u.h[0] = *(const v8bf*)(ap);
                u.h[1] = *(const v8bf*)(ap + 16);
                af[i] = u.v;
            }
            // ---- B fragments: B[k,n] = W[n,k]; same K layout as A
            v16bf bfv[4];
            #pragma unroll
            for (int j = 0; j < 4; j++) {
                const bf16* bp = bT + (wn * 64 + j * 16 + l16) * TROW_ + half * 8;
                union { v16bf v; v8bf h[2]; } u;
                u.h[0] = *(const v8bf*)(bp);
                u.h[1] = *(const v8bf*)(bp + 16);
                bfv[j] = u.v;
            }
            #pragma unroll
            for (int i = 0; i < 2; i++)
                #pragma unroll
                for (int j = 0; j < 4; j++)
                    acc[i][j] = __builtin_amdgcn_wmma_f32_16x16x32_bf16(
                        false, af[i], false, bfv[j], (short)0, acc[i][j], false, false);
        }
    }
    if (!active) return;

    // ---- epilogue: bias + activation + residual; f32 and/or bf16 stores ----
    #pragma unroll
    for (int i = 0; i < 2; i++) {
        #pragma unroll
        for (int j = 0; j < 4; j++) {
            const int col = n0 + j * 16 + l16;
            const float bs = bias ? bias[col] : 0.f;
            #pragma unroll
            for (int r = 0; r < 8; r++) {
                const int row = m0 + i * 16 + half * 8 + r;  // C/D layout: VGPR r -> M = half*8+r
                float v = acc[i][j][r] + bs;
                if (act == 1) {                               // relu
                    v = v > 0.f ? v : 0.f;
                } else if (act == 2) {                        // tanh-gelu
                    float x3 = v * v * v;
                    v = 0.5f * v * (1.f + tanhf(0.7978845608f * (v + 0.044715f * x3)));
                } else if (act == 3) {                        // sigmoid
                    v = 1.f / (1.f + __expf(-v));
                }
                const long idx = cOff + (long)row * ldc + col;
                if (resid) v += resid[idx];
                if (C)  C[idx]  = v;
                if (Cb) Cb[idx] = (bf16)v;
            }
        }
    }
}

// =====================================================================
// LayerNorm over D=1024, one block per row, f32 and/or bf16 outputs.
// =====================================================================
__global__ __launch_bounds__(256) void ln_kernel(
    const float* __restrict__ X, const float* __restrict__ w, const float* __restrict__ b,
    float* __restrict__ outF, bf16* __restrict__ outB, float eps)
{
    __shared__ float red[256];
    const int row = blockIdx.x;
    const int t = threadIdx.x;
    const float* xr = X + (long)row * D_;
    float v[4], s = 0.f;
    #pragma unroll
    for (int i = 0; i < 4; i++) { v[i] = xr[t + i * 256]; s += v[i]; }
    red[t] = s; __syncthreads();
    for (int o = 128; o > 0; o >>= 1) { if (t < o) red[t] += red[t + o]; __syncthreads(); }
    const float mean = red[0] * (1.f / D_);
    __syncthreads();
    s = 0.f;
    #pragma unroll
    for (int i = 0; i < 4; i++) { float d = v[i] - mean; s += d * d; }
    red[t] = s; __syncthreads();
    for (int o = 128; o > 0; o >>= 1) { if (t < o) red[t] += red[t + o]; __syncthreads(); }
    const float rstd = rsqrtf(red[0] * (1.f / D_) + eps);
    #pragma unroll
    for (int i = 0; i < 4; i++) {
        const int c = t + i * 256;
        const float y = (v[i] - mean) * rstd * w[c] + b[c];
        if (outF) outF[(long)row * D_ + c] = y;
        if (outB) outB[(long)row * D_ + c] = (bf16)y;
    }
}

// =====================================================================
// Softmax over K=1024: optional rel-pos gathered bias, 1/8 scale, mask,
// f32 probs written in place (d_out attn map) + bf16 copy for attn@V.
// grid = (q=1024, z=32); block = 256.
// =====================================================================
__global__ __launch_bounds__(256) void softmax_kernel(
    float* __restrict__ S, const float* __restrict__ attn2,
    const int* __restrict__ rpm, const int* __restrict__ mask,
    bf16* __restrict__ P, int useBias)
{
    __shared__ float red[256];
    __shared__ float a2[PK_];
    const int q = blockIdx.x, z = blockIdx.y, b = z >> 4;
    const int t = threadIdx.x;
    if (useBias && t < PK_) a2[t] = attn2[((long)z * L_ + q) * PK_ + t];
    __syncthreads();
    float* Sr = S + (long)z * SQ_ + (long)q * L_;
    const int* rr = useBias ? (rpm + ((long)b * L_ + q) * L_) : nullptr;
    const int* mr = mask + b * L_;
    float v[4], mx = -3.4e38f;
    #pragma unroll
    for (int i = 0; i < 4; i++) {
        const int k = t + i * 256;
        float val = Sr[k];
        if (useBias) val += a2[rr[k]];
        val *= 0.125f;                       // 1/sqrt(DK)
        if (mr[k] == 0) val = -9e9f;
        v[i] = val; mx = fmaxf(mx, val);
    }
    red[t] = mx; __syncthreads();
    for (int o = 128; o > 0; o >>= 1) { if (t < o) red[t] = fmaxf(red[t], red[t + o]); __syncthreads(); }
    mx = red[0]; __syncthreads();
    float s = 0.f;
    #pragma unroll
    for (int i = 0; i < 4; i++) { v[i] = __expf(v[i] - mx); s += v[i]; }
    red[t] = s; __syncthreads();
    for (int o = 128; o > 0; o >>= 1) { if (t < o) red[t] += red[t + o]; __syncthreads(); }
    const float inv = 1.f / red[0];
    #pragma unroll
    for (int i = 0; i < 4; i++) {
        const int k = t + i * 256;
        const float p = v[i] * inv;
        Sr[k] = p;
        P[(long)z * SQ_ + (long)q * L_ + k] = (bf16)p;
    }
}

// out[z,q,k] = mask[b,k] ? 1 : 0  (broadcast mask map)
__global__ __launch_bounds__(256) void fill_mask_kernel(
    float* __restrict__ out, const int* __restrict__ mask)
{
    long i = ((long)blockIdx.x * 256 + threadIdx.x) * 4;
    #pragma unroll
    for (int u = 0; u < 4; u++, i++) {
        const int z = (int)(i >> 20);
        const int k = (int)(i & (L_ - 1));
        const int b = z >> 4;
        out[i] = mask[b * L_ + k] ? 1.f : 0.f;
    }
}

// attn2[b,h,q,p] = q1[b,h,q,:] . rel_k[p,:]
__global__ __launch_bounds__(256) void relbias_kernel(
    const float* __restrict__ q, const float* __restrict__ relk, float* __restrict__ out)
{
    const int idx = blockIdx.x * 256 + threadIdx.x;  // ZN_*L_*PK_ = 163840
    if (idx >= ZN_ * L_ * PK_) return;
    const int p = idx % PK_;
    const int qq = (idx / PK_) & (L_ - 1);
    const int z = idx / (PK_ * L_);
    const int b = z >> 4, h = z & 15;
    const float* qp = q + (long)b * ((long)L_ * D_) + (long)qq * D_ + h * DK_;
    const float* rp = relk + p * DK_;
    float s = 0.f;
    #pragma unroll
    for (int d = 0; d < DK_; d++) s += qp[d] * rp[d];
    out[idx] = s;
}

// vT[b,h,d,l] = (bf16) v[b,l,h*DK+d]
__global__ __launch_bounds__(256) void vtrans_kernel(
    const float* __restrict__ v, bf16* __restrict__ vt)
{
    const long i = (long)blockIdx.x * 256 + threadIdx.x;  // 2,097,152
    const int l = (int)(i & (L_ - 1));
    const int d = (int)((i >> 10) & (DK_ - 1));
    const int z = (int)(i >> 16);
    const int b = z >> 4, h = z & 15;
    vt[i] = (bf16)v[(long)b * ((long)L_ * D_) + (long)l * D_ + h * DK_ + d];
}

__global__ __launch_bounds__(256) void f2bf_kernel(
    const float* __restrict__ in, bf16* __restrict__ out, int n)
{
    const int i = blockIdx.x * 256 + threadIdx.x;
    if (i < n) out[i] = (bf16)in[i];
}

// Gm = (bf16)(O * G)   (gate-multiply before output projection)
__global__ __launch_bounds__(256) void gatemul_kernel(
    const float* __restrict__ o, const float* __restrict__ g, bf16* __restrict__ out)
{
    const long i = (long)blockIdx.x * 256 + threadIdx.x;
    out[i] = (bf16)(o[i] * g[i]);
}

// =====================================================================
// Host orchestration
// =====================================================================
extern "C" void kernel_launch(void* const* d_in, const int* in_sizes, int n_in,
                              void* d_out, int out_size, void* d_ws, size_t ws_size,
                              hipStream_t stream)
{
    const float* src   = (const float*)d_in[0];
    const float* tgt   = (const float*)d_in[1];
    const int*   rpm   = (const int*)d_in[2];
    const int*   smask = (const int*)d_in[3];
    const int*   tmask = (const int*)d_in[4];
    auto F = [&](int i) { return (const float*)d_in[i]; };
    // params (flattened dict order): 5..17 sa, 18..29 ca, 30..39 tail
    const float* sa_ln_w = F(5),  *sa_ln_b = F(6);
    const float* sa_vb1 = F(10), *sa_vb2 = F(12), *sa_gb = F(14), *sa_ob = F(16);
    const float* rel_k  = F(17);
    const float* ca_ln_w = F(18), *ca_ln_b = F(19);
    const float* ca_vb1 = F(23), *ca_vb2 = F(25), *ca_gb = F(27), *ca_ob = F(29);
    const float* n1w = F(30), *n1b = F(31), *n2w = F(32), *n2b = F(33), *n3w = F(34), *n3b = F(35);
    const float* ff_b1 = F(37), *ff_b2 = F(39);
    const int wsrc[14] = {7, 8, 9, 11, 13, 15, 20, 21, 22, 24, 26, 28, 36, 38};
    // wbf index: 0 wq,1 wk,2 vw1,3 vw2,4 gw,5 ow (sa); 6..11 (ca); 12 ffw1, 13 ffw2

    float* out = (float*)d_out;

    // ---- workspace carve-up (deterministic bump allocator) ----
    char* base = (char*)d_ws;
    size_t off = 0;
    auto alloc = [&](size_t bytes) { void* p = base + off; off += (bytes + 255) & ~(size_t)255; return p; };
    bf16* wbf[14];
    for (int i = 0; i < 14; i++) wbf[i] = (bf16*)alloc((size_t)D_ * D_ * 2);
    float* F_xn1 = (float*)alloc(LD_ * 4);
    float* F_x1  = (float*)alloc(LD_ * 4);
    float* F_x2  = (float*)alloc(LD_ * 4);
    float* F_xn2 = (float*)alloc(LD_ * 4);
    float* F_q   = (float*)alloc(LD_ * 4);
    float* F_v   = (float*)alloc(LD_ * 4);
    float* F_O   = (float*)alloc(LD_ * 4);
    float* F_G   = (float*)alloc(LD_ * 4);
    bf16* B_sn  = (bf16*)alloc(LD_ * 2);
    bf16* B_q   = (bf16*)alloc(LD_ * 2);
    bf16* B_k   = (bf16*)alloc(LD_ * 2);
    bf16* B_tmp = (bf16*)alloc(LD_ * 2);
    bf16* B_gm  = (bf16*)alloc(LD_ * 2);
    bf16* B_tc  = (bf16*)alloc(LD_ * 2);
    bf16* B_vT  = (bf16*)alloc((size_t)ZN_ * VT_ * 2);
    float* F_a2 = (float*)alloc((size_t)ZN_ * L_ * PK_ * 4);
    bf16* P_bf  = (bf16*)alloc((size_t)ATT_ * 2);

    const dim3 blk(256);
    auto gemm = [&](const bf16* A, const bf16* W, float* C, bf16* Cb,
                    const float* bias, const float* resid,
                    int M, int N, int K, int lda, int ldb, int ldc,
                    long aO, long aI, long wO, long wI, long cO, long cI,
                    int zInner, int nz, int act) {
        dim3 g((N + 127) / 128, (M + 127) / 128, nz);
        gemm_bf16_kernel<<<g, blk, 0, stream>>>(A, W, C, Cb, bias, resid,
                                                M, N, K, lda, ldb, ldc,
                                                aO, aI, wO, wI, cO, cI, zInner, act);
    };
    const int R = B_ * L_;           // 2048 rows for fused-batch GEMMs
    const long LDl = LD_;

    // 0) weights f32 -> bf16
    for (int i = 0; i < 14; i++)
        f2bf_kernel<<<(D_ * D_) / 256, blk, 0, stream>>>(F(wsrc[i]), wbf[i], D_ * D_);

    // ================= self-attention =================
    ln_kernel<<<R, blk, 0, stream>>>(src,   n1w, n1b, F_xn1, nullptr, 1e-5f);     // norm1
    ln_kernel<<<R, blk, 0, stream>>>(F_xn1, sa_ln_w, sa_ln_b, nullptr, B_sn, 1e-6f); // mha ln
    gemm(B_sn, wbf[0], F_q, B_q, nullptr, nullptr, R, D_, D_, D_, D_, D_, 0,0,0,0,0,0, 1, 1, 0); // q
    gemm(B_sn, wbf[1], nullptr, B_k, nullptr, nullptr, R, D_, D_, D_, D_, D_, 0,0,0,0,0,0, 1, 1, 0); // k
    gemm(B_sn, wbf[2], nullptr, B_tmp, sa_vb1, nullptr, R, D_, D_, D_, D_, D_, 0,0,0,0,0,0, 1, 1, 1); // relu(v1)
    gemm(B_tmp, wbf[3], F_v, nullptr, sa_vb2, nullptr, R, D_, D_, D_, D_, D_, 0,0,0,0,0,0, 1, 1, 0);  // v
    vtrans_kernel<<<LD_ / 256, blk, 0, stream>>>(F_v, B_vT);
    relbias_kernel<<<(ZN_ * L_ * PK_ + 255) / 256, blk, 0, stream>>>(F_q, rel_k, F_a2);
    // logits: S[b,h,q,k] = q1.k1  (straight into d_out as_s region)
    gemm(B_q, B_k, out + OFF_ASS, nullptr, nullptr, nullptr,
         L_, L_, DK_, D_, D_, L_,
         LDl, DK_, LDl, DK_, (long)H_ * SQ_, SQ_, H_, ZN_, 0);
    softmax_kernel<<<dim3(L_, ZN_), blk, 0, stream>>>(out + OFF_ASS, F_a2, rpm, smask, P_bf, 1);
    fill_mask_kernel<<<(int)(ATT_ / 1024), blk, 0, stream>>>(out + OFF_AMS, smask);
    // attn @ V -> O[b,q,h*DK+d]
    gemm(P_bf, B_vT, F_O, nullptr, nullptr, nullptr,
         L_, DK_, L_, L_, L_, D_,
         (long)H_ * SQ_, SQ_, (long)H_ * VT_, VT_, LDl, DK_, H_, ZN_, 0);
    gemm(B_sn, wbf[4], F_G, nullptr, sa_gb, nullptr, R, D_, D_, D_, D_, D_, 0,0,0,0,0,0, 1, 1, 3); // sigmoid gate
    gatemul_kernel<<<LD_ / 256, blk, 0, stream>>>(F_O, F_G, B_gm);
    gemm(B_gm, wbf[5], F_x1, nullptr, sa_ob, src, R, D_, D_, D_, D_, D_, 0,0,0,0,0,0, 1, 1, 0);   // out proj + resid

    // ================= cross-attention =================
    ln_kernel<<<R, blk, 0, stream>>>(F_x1,  n2w, n2b, F_xn2, nullptr, 1e-5f);        // norm2
    ln_kernel<<<R, blk, 0, stream>>>(F_xn2, ca_ln_w, ca_ln_b, nullptr, B_sn, 1e-6f); // s (keys/gate)
    ln_kernel<<<R, blk, 0, stream>>>(tgt,   ca_ln_w, ca_ln_b, nullptr, B_tc, 1e-6f); // t (queries/values)
    gemm(B_tc, wbf[6], nullptr, B_q, nullptr, nullptr, R, D_, D_, D_, D_, D_, 0,0,0,0,0,0, 1, 1, 0); // q (tgt)
    gemm(B_sn, wbf[7], nullptr, B_k, nullptr, nullptr, R, D_, D_, D_, D_, D_, 0,0,0,0,0,0, 1, 1, 0); // k (src)
    gemm(B_tc, wbf[8], nullptr, B_tmp, ca_vb1, nullptr, R, D_, D_, D_, D_, D_, 0,0,0,0,0,0, 1, 1, 1);
    gemm(B_tmp, wbf[9], F_v, nullptr, ca_vb2, nullptr, R, D_, D_, D_, D_, D_, 0,0,0,0,0,0, 1, 1, 0);
    vtrans_kernel<<<LD_ / 256, blk, 0, stream>>>(F_v, B_vT);
    // transposed logits: S_c[b,h,i in Ls, j in Lt] = k1[i].q1[j]
    gemm(B_k, B_q, out + OFF_ASC, nullptr, nullptr, nullptr,
         L_, L_, DK_, D_, D_, L_,
         LDl, DK_, LDl, DK_, (long)H_ * SQ_, SQ_, H_, ZN_, 0);
    softmax_kernel<<<dim3(L_, ZN_), blk, 0, stream>>>(out + OFF_ASC, nullptr, nullptr, tmask, P_bf, 0);
    fill_mask_kernel<<<(int)(ATT_ / 1024), blk, 0, stream>>>(out + OFF_AMC, tmask);
    gemm(P_bf, B_vT, F_O, nullptr, nullptr, nullptr,
         L_, DK_, L_, L_, L_, D_,
         (long)H_ * SQ_, SQ_, (long)H_ * VT_, VT_, LDl, DK_, H_, ZN_, 0);
    gemm(B_sn, wbf[10], F_G, nullptr, ca_gb, nullptr, R, D_, D_, D_, D_, D_, 0,0,0,0,0,0, 1, 1, 3);
    gatemul_kernel<<<LD_ / 256, blk, 0, stream>>>(F_O, F_G, B_gm);
    gemm(B_gm, wbf[11], F_x2, nullptr, ca_ob, F_x1, R, D_, D_, D_, D_, D_, 0,0,0,0,0,0, 1, 1, 0);

    // ================= FFN =================
    ln_kernel<<<R, blk, 0, stream>>>(F_x2, n3w, n3b, nullptr, B_sn, 1e-5f);          // norm3
    gemm(B_sn, wbf[12], nullptr, B_tmp, ff_b1, nullptr, R, D_, D_, D_, D_, D_, 0,0,0,0,0,0, 1, 1, 2); // gelu
    gemm(B_tmp, wbf[13], out + OFF_X, nullptr, ff_b2, F_x2, R, D_, D_, D_, D_, D_, 0,0,0,0,0,0, 1, 1, 0);
}
